// cylinder_fea_78211354460783
// MI455X (gfx1250) — compile-verified, matrix operand order checked by reading, not run on hardware
//
#include <hip/hip_runtime.h>
#include <hip/hip_bf16.h>

// ---------------------------------------------------------------------------
// cylinder_fea for MI455X (gfx1250, wave32, WMMA).
// Dense GEMMs (K in {64,128,256}) run through v_wmma_f32_16x16x32_f16 with
// 32x32 register blocking per wave: 2 A-frags x 2 B-frags -> 4 accumulators,
// 4 WMMAs per K-step (2x arithmetic intensity vs 16x16 tiles, and 4
// independent accumulator chains to hide WMMA RAW latency).
// Pooling reproduces jnp.unique(sorted)+segment_max via dense per-key
// atomic-max slots (monotone float<->uint encoding) + an in-LDS scan.
// ---------------------------------------------------------------------------

#define LL     4
#define DD     256
#define HH     4
#define GYV    200
#define NKEYS  (GYV * GYV)

typedef _Float16 h16;
typedef __attribute__((ext_vector_type(16))) _Float16 v16h;
typedef __attribute__((ext_vector_type(8)))  _Float16 v8h;
typedef __attribute__((ext_vector_type(8)))  float    v8f;

union HFrag { v16h v; v8h h[2]; };

static __device__ __forceinline__ unsigned encF(float f) {
  unsigned u = __float_as_uint(f);
  return (u & 0x80000000u) ? ~u : (u | 0x80000000u);   // monotone order-preserving map
}
static __device__ __forceinline__ float decF(unsigned u) {
  u = (u & 0x80000000u) ? (u & 0x7fffffffu) : ~u;
  return __uint_as_float(u);
}

// ---------------------------------------------------------------------------
// WMMA GEMM: out[M,Nc] = act( (A[M,K] x Bt[Nc,K]^T + bias) * scale + shift )
// A, Bt are f16 row-major. One wave owns a (16*MT)x(16*NT) tile; K is a
// compile-time multiple of 32 so the loop unrolls into (K/32)*MT*NT WMMAs.
// Fragment layout (CDNA5 16-bit A-matrix): lanes 0-15 hold K 0-7 / 16-23,
// lanes 16-31 hold K 8-15 / 24-31 -> two contiguous 16B v8h loads per lane.
// ---------------------------------------------------------------------------
template <int K, int MT, int NT>
__global__ __launch_bounds__(256) void gemm_wmma_f16(
    const h16* __restrict__ A, const h16* __restrict__ Bt,
    const float* __restrict__ bias,
    const float* __restrict__ scale, const float* __restrict__ shift,
    float* __restrict__ outF, h16* __restrict__ outH,
    int M, int Nc, int doRelu)
{
  const int  lane   = threadIdx.x & 31;
  const int  wave   = threadIdx.x >> 5;
  const int  nTiles = Nc / (16 * NT);
  const long mTiles = (long)((M + 16 * MT - 1) / (16 * MT));
  const long tile   = (long)blockIdx.x * 8 + wave;
  if (tile >= mTiles * (long)nTiles) return;          // wave-uniform: EXEC stays full
  const int  nT = (int)(tile % nTiles);
  const long mT = tile / nTiles;

  const int halfSel = lane >> 4;                      // lanes 0-15 vs 16-31 (K halves)
  const h16* pa[MT];
#pragma unroll
  for (int mi = 0; mi < MT; ++mi) {
    int r = (int)(mT * 16 * MT) + mi * 16 + (lane & 15);
    if (r >= M) r = M - 1;                            // clamp loads; stores masked below
    pa[mi] = A + (long)r * K + halfSel * 8;
  }
  const h16* pb[NT];
#pragma unroll
  for (int ni = 0; ni < NT; ++ni)
    pb[ni] = Bt + (long)(nT * 16 * NT + ni * 16 + (lane & 15)) * K + halfSel * 8;

  v8f acc[MT][NT];
#pragma unroll
  for (int mi = 0; mi < MT; ++mi)
#pragma unroll
    for (int ni = 0; ni < NT; ++ni) acc[mi][ni] = (v8f){};

#pragma unroll
  for (int kb = 0; kb < K; kb += 32) {
    HFrag fa[MT], fb[NT];
#pragma unroll
    for (int mi = 0; mi < MT; ++mi) {
      fa[mi].h[0] = *(const v8h*)(pa[mi] + kb);       // K = kb + sel*8 .. +7
      fa[mi].h[1] = *(const v8h*)(pa[mi] + kb + 16);  // K = kb + 16 + sel*8 .. +7
    }
#pragma unroll
    for (int ni = 0; ni < NT; ++ni) {
      fb[ni].h[0] = *(const v8h*)(pb[ni] + kb);
      fb[ni].h[1] = *(const v8h*)(pb[ni] + kb + 16);
    }
    if (kb + 32 < K) {                                // global_prefetch_b8 next K-block
      __builtin_prefetch((const void*)(pa[0] + kb + 32), 0, 1);
      __builtin_prefetch((const void*)(pb[0] + kb + 32), 0, 1);
    }
#pragma unroll
    for (int mi = 0; mi < MT; ++mi)
#pragma unroll
      for (int ni = 0; ni < NT; ++ni)
        acc[mi][ni] = __builtin_amdgcn_wmma_f32_16x16x32_f16(
            false, fa[mi].v, false, fb[ni].v, (short)0, acc[mi][ni], false, false);
  }

  // C/D layout: col = lane%16, rows = tileRow + (lane<16?0:8) + r
#pragma unroll
  for (int ni = 0; ni < NT; ++ni) {
    const int col = nT * 16 * NT + ni * 16 + (lane & 15);
    const float bc = bias  ? bias[col]  : 0.f;
    const float sc = scale ? scale[col] : 1.f;
    const float sh = shift ? shift[col] : 0.f;
#pragma unroll
    for (int mi = 0; mi < MT; ++mi) {
      const long rowBase = mT * 16 * MT + mi * 16 + halfSel * 8;
#pragma unroll
      for (int r = 0; r < 8; ++r) {
        const long row = rowBase + r;
        if (row < M) {
          float x = (acc[mi][ni][r] + bc) * sc + sh;
          if (doRelu) x = fmaxf(x, 0.f);
          if (outF) outF[row * Nc + col] = x;
          if (outH) outH[row * Nc + col] = (h16)x;
        }
      }
    }
  }
}

// --- prep: W[K,Nc] f32 -> Wt[Nc,K] f16 (so B fragments load contiguously) --
__global__ void wt_convert(const float* __restrict__ W, h16* __restrict__ Wt,
                           int K, int Nc)
{
  int idx = blockIdx.x * blockDim.x + threadIdx.x;
  if (idx >= K * Nc) return;
  int nc = idx / K, k = idx % K;
  Wt[idx] = (h16)W[k * Nc + nc];
}

// --- prep: fold eval-BN into per-column scale/shift ------------------------
__global__ void bn_fold(const float* __restrict__ g, const float* __restrict__ b,
                        const float* __restrict__ m, const float* __restrict__ v,
                        float* __restrict__ scale, float* __restrict__ shift, int d)
{
  int i = blockIdx.x * blockDim.x + threadIdx.x;
  if (i >= d) return;
  float s = g[i] * rsqrtf(v[i] + 1e-5f);
  scale[i] = s;
  shift[i] = b[i] - m[i] * s;
}

__global__ void cvt_f2h(const float* __restrict__ src, h16* __restrict__ dst, long cnt)
{
  long i = (long)blockIdx.x * blockDim.x + threadIdx.x;
  if (i < cnt) dst[i] = (h16)src[i];
}

// --- stage 1: BN0(pt) -> Lin(3->64) -> BN1 -> ReLU (K=3 too small for WMMA)
__global__ void pp_stage1(const float* __restrict__ pt,
    const float* __restrict__ g0, const float* __restrict__ b0,
    const float* __restrict__ m0, const float* __restrict__ v0,
    const float* __restrict__ W1, const float* __restrict__ bb1,
    const float* __restrict__ g1, const float* __restrict__ b1,
    const float* __restrict__ m1, const float* __restrict__ v1,
    h16* __restrict__ h1, int n)
{
  int i = blockIdx.x * blockDim.x + threadIdx.x;
  if (i >= n) return;
  float x0 = (pt[i*3+0] - m0[0]) * rsqrtf(v0[0] + 1e-5f) * g0[0] + b0[0];
  float x1 = (pt[i*3+1] - m0[1]) * rsqrtf(v0[1] + 1e-5f) * g0[1] + b0[1];
  float x2 = (pt[i*3+2] - m0[2]) * rsqrtf(v0[2] + 1e-5f) * g0[2] + b0[2];
  for (int j = 0; j < 64; ++j) {
    float t = bb1[j] + x0 * W1[j] + x1 * W1[64 + j] + x2 * W1[128 + j];
    t = (t - m1[j]) * rsqrtf(v1[j] + 1e-5f) * g1[j] + b1[j];
    h1[(long)i * 64 + j] = (h16)fmaxf(t, 0.f);
  }
}

// --- attention: per (point, head); L=4 keys, dh=64 -------------------------
__global__ void mha_attn(const float* __restrict__ q,
                         const h16* __restrict__ k,
                         const h16* __restrict__ v,
                         h16* __restrict__ o, int n)
{
  int idx = blockIdx.x * blockDim.x + threadIdx.x;
  if (idx >= n * HH) return;
  int pt = idx >> 2, h = idx & 3;
  const float* qp = q + (long)pt * DD + h * 64;
  const h16*   kp = k + (long)pt * LL * DD + h * 64;
  float s0 = 0, s1 = 0, s2 = 0, s3 = 0;
  for (int d = 0; d < 64; ++d) {
    float qd = qp[d];
    s0 += qd * (float)kp[d];
    s1 += qd * (float)kp[DD + d];
    s2 += qd * (float)kp[2 * DD + d];
    s3 += qd * (float)kp[3 * DD + d];
  }
  s0 *= 0.125f; s1 *= 0.125f; s2 *= 0.125f; s3 *= 0.125f;   // 1/sqrt(64)
  float mx = fmaxf(fmaxf(s0, s1), fmaxf(s2, s3));
  float e0 = __expf(s0 - mx), e1 = __expf(s1 - mx);
  float e2 = __expf(s2 - mx), e3 = __expf(s3 - mx);
  float inv = 1.f / (e0 + e1 + e2 + e3);
  e0 *= inv; e1 *= inv; e2 *= inv; e3 *= inv;
  const h16* vp = v + (long)pt * LL * DD + h * 64;
  h16*       op = o + (long)pt * DD + h * 64;
  for (int d = 0; d < 64; ++d) {
    float r = e0 * (float)vp[d] + e1 * (float)vp[DD + d]
            + e2 * (float)vp[2 * DD + d] + e3 * (float)vp[3 * DD + d];
    op[d] = (h16)r;
  }
}

// --- pooling: dense per-key atomic max -------------------------------------
__global__ void pool_atomic(const int* __restrict__ xy,
    const float* __restrict__ mlp, const float* __restrict__ sam,
    const float* __restrict__ pix,
    unsigned* __restrict__ slotM, unsigned* __restrict__ slotS,
    unsigned* __restrict__ slotP, unsigned* __restrict__ counts, int n)
{
  long idx = (long)blockIdx.x * blockDim.x + threadIdx.x;
  if (idx >= (long)n * DD) return;
  int pt = (int)(idx >> 8), c = (int)(idx & 255);
  int key = xy[2 * pt] * GYV + xy[2 * pt + 1];
  if (c == 0) atomicAdd(&counts[key], 1u);
  long s = (long)key * DD + c;
  atomicMax(&slotM[s], encF(mlp[idx]));
  atomicMax(&slotS[s], encF(sam[idx]));
  atomicMax(&slotP[s], encF(pix[idx]));
}

// --- exclusive scan of presence flags -> sorted-unique ranks ---------------
__global__ void scan_ranks(const unsigned* __restrict__ counts,
                           unsigned* __restrict__ rank,
                           unsigned* __restrict__ keyOfRank,
                           unsigned* __restrict__ nuniq)
{
  __shared__ unsigned buf[1024];
  __shared__ unsigned carry;
  if (threadIdx.x == 0) carry = 0;
  __syncthreads();
  for (int base = 0; base < NKEYS; base += 1024) {
    int i = base + (int)threadIdx.x;
    unsigned f = (i < NKEYS && counts[i] > 0u) ? 1u : 0u;
    buf[threadIdx.x] = f;
    __syncthreads();
    for (int off = 1; off < 1024; off <<= 1) {
      unsigned t = (threadIdx.x >= (unsigned)off) ? buf[threadIdx.x - off] : 0u;
      __syncthreads();
      buf[threadIdx.x] += t;
      __syncthreads();
    }
    unsigned incl = buf[threadIdx.x];
    unsigned r = carry + incl - f;                    // exclusive rank
    if (i < NKEYS) {
      rank[i] = r;
      if (f) keyOfRank[r] = (unsigned)i;
    }
    __syncthreads();
    if (threadIdx.x == 1023) carry += incl;
    __syncthreads();
  }
  if (threadIdx.x == 1023) *nuniq = carry;
}

__global__ void write_inv(const int* __restrict__ xy,
                          const unsigned* __restrict__ rank,
                          float* __restrict__ o_inv, int n)
{
  int i = blockIdx.x * blockDim.x + threadIdx.x;
  if (i >= n) return;
  int key = xy[2 * i] * GYV + xy[2 * i + 1];
  o_inv[i] = (float)rank[key];
}

// --- gather pooled rows in rank order, emit unq/seg/pix, feed final GEMMs --
__global__ void pool_gather(
    const unsigned* __restrict__ slotM, const unsigned* __restrict__ slotS,
    const unsigned* __restrict__ slotP,
    const unsigned* __restrict__ keyOfRank, const unsigned* __restrict__ nuniq,
    float* __restrict__ pmlp, h16* __restrict__ pmlp_h,
    float* __restrict__ o_seg, float* __restrict__ o_pix,
    float* __restrict__ o_unq, int n)
{
  long idx = (long)blockIdx.x * blockDim.x + threadIdx.x;
  if (idx >= (long)n * DD) return;
  int r = (int)(idx >> 8), c = (int)(idx & 255);
  unsigned U = *nuniq;
  float pm = 0.f, ps = 0.f, pp = 0.f;
  int key = NKEYS;                                    // jnp.unique fill_value = GY*GY
  if ((unsigned)r < U) {
    key = (int)keyOfRank[r];
    long s = (long)key * DD + c;
    pm = decF(slotM[s]); ps = decF(slotS[s]); pp = decF(slotP[s]);
  }
  pmlp[idx]  = pm;
  pmlp_h[idx] = (h16)pm;
  o_seg[idx] = ps;
  o_pix[idx] = pp;
  if (c < 3) {
    float u = (c == 0) ? 0.f : (c == 1) ? (float)(key / GYV) : (float)(key % GYV);
    o_unq[(long)r * 3 + c] = u;
  }
}

// ---------------------------------------------------------------------------
extern "C" void kernel_launch(void* const* d_in, const int* in_sizes, int n_in,
                              void* d_out, int out_size, void* d_ws, size_t ws_size,
                              hipStream_t stream)
{
  (void)n_in; (void)out_size; (void)ws_size;
  const int  n  = in_sizes[0] / 3;                    // N points
  const long nL = (long)n;

  // inputs in setup_inputs() dict order
  const float* pt_fea = (const float*)d_in[0];
  const int*   xy     = (const int*)  d_in[1];
  const float* segfea = (const float*)d_in[2];
  const float* pixfea = (const float*)d_in[3];
  const float *bn0_g=(const float*)d_in[4],  *bn0_b=(const float*)d_in[5],
              *bn0_m=(const float*)d_in[6],  *bn0_v=(const float*)d_in[7];
  const float *bn1_g=(const float*)d_in[8],  *bn1_b=(const float*)d_in[9],
              *bn1_m=(const float*)d_in[10], *bn1_v=(const float*)d_in[11];
  const float *bn2_g=(const float*)d_in[12], *bn2_b=(const float*)d_in[13],
              *bn2_m=(const float*)d_in[14], *bn2_v=(const float*)d_in[15];
  const float *bn3_g=(const float*)d_in[16], *bn3_b=(const float*)d_in[17],
              *bn3_m=(const float*)d_in[18], *bn3_v=(const float*)d_in[19];
  const float *W1=(const float*)d_in[20], *b1=(const float*)d_in[21];
  const float *W2=(const float*)d_in[22], *b2=(const float*)d_in[23];
  const float *W3=(const float*)d_in[24], *b3=(const float*)d_in[25];
  const float *W4=(const float*)d_in[26], *b4=(const float*)d_in[27];
  const float *Wq=(const float*)d_in[28], *bq=(const float*)d_in[29];
  const float *Wk=(const float*)d_in[30], *bk=(const float*)d_in[31];
  const float *Wv=(const float*)d_in[32], *bv=(const float*)d_in[33];
  const float *Wo=(const float*)d_in[34], *bo=(const float*)d_in[35];
  const float *W128=(const float*)d_in[36], *b128=(const float*)d_in[37];
  const float *Wc=(const float*)d_in[38],  *bc=(const float*)d_in[39];

  // outputs, concatenated flat in return order (stored as float)
  float* out    = (float*)d_out;
  float* o_unq  = out;
  float* o_proc = o_unq  + nL * 3;
  float* o_seg  = o_proc + nL * 16;
  float* o_pix  = o_seg  + nL * DD;
  float* o_inv  = o_pix  + nL * DD;

  // workspace bump allocator (256B aligned)
  char* wp = (char*)d_ws;
  auto alloc = [&](size_t bytes) -> void* {
    void* p = (void*)wp;
    wp += (bytes + 255) & ~(size_t)255;
    return p;
  };
  h16* W2t   = (h16*)alloc(sizeof(h16) * 128 * 64);
  h16* W3t   = (h16*)alloc(sizeof(h16) * 256 * 128);
  h16* W4t   = (h16*)alloc(sizeof(h16) * 256 * 256);
  h16* Wqt   = (h16*)alloc(sizeof(h16) * 256 * 256);
  h16* Wkt   = (h16*)alloc(sizeof(h16) * 256 * 256);
  h16* Wvt   = (h16*)alloc(sizeof(h16) * 256 * 256);
  h16* Wot   = (h16*)alloc(sizeof(h16) * 256 * 256);
  h16* W128t = (h16*)alloc(sizeof(h16) * 128 * 256);
  h16* Wct   = (h16*)alloc(sizeof(h16) * 16 * 128);
  float* sc2 = (float*)alloc(sizeof(float) * 128);
  float* sh2 = (float*)alloc(sizeof(float) * 128);
  float* sc3 = (float*)alloc(sizeof(float) * 256);
  float* sh3 = (float*)alloc(sizeof(float) * 256);
  h16* h1     = (h16*)alloc(sizeof(h16) * nL * 64);
  h16* h2     = (h16*)alloc(sizeof(h16) * nL * 128);
  h16* h3     = (h16*)alloc(sizeof(h16) * nL * 256);
  float* mlp  = (float*)alloc(sizeof(float) * nL * 256);
  h16* mlp_h  = (h16*)alloc(sizeof(h16) * nL * 256);
  float* qb   = (float*)alloc(sizeof(float) * nL * 256);
  h16* seg_h  = (h16*)alloc(sizeof(h16) * nL * LL * 256);
  h16* k_h    = (h16*)alloc(sizeof(h16) * nL * LL * 256);
  h16* v_h    = (h16*)alloc(sizeof(h16) * nL * LL * 256);
  h16* attn_h = (h16*)alloc(sizeof(h16) * nL * 256);
  float* sam  = (float*)alloc(sizeof(float) * nL * 256);
  unsigned* counts    = (unsigned*)alloc(sizeof(unsigned) * NKEYS);
  unsigned* rankArr   = (unsigned*)alloc(sizeof(unsigned) * NKEYS);
  unsigned* keyOfRank = (unsigned*)alloc(sizeof(unsigned) * NKEYS);
  unsigned* nuniq     = (unsigned*)alloc(sizeof(unsigned));
  unsigned* slots     = (unsigned*)alloc(sizeof(unsigned) * 3 * (size_t)NKEYS * DD);
  unsigned* slotM = slots;
  unsigned* slotS = slots + (size_t)NKEYS * DD;
  unsigned* slotP = slots + 2 * (size_t)NKEYS * DD;
  float* pmlp  = (float*)alloc(sizeof(float) * nL * 256);
  h16* pmlp_h  = (h16*)alloc(sizeof(h16) * nL * 256);
  h16* ori_h   = (h16*)alloc(sizeof(h16) * nL * 128);

  auto cd = [](long a, long b) { return (int)((a + b - 1) / b); };
  auto gemmGrid = [&](long M, int Nc, int MT, int NT) {
    long tiles = ((M + 16 * MT - 1) / (16 * MT)) * (long)(Nc / (16 * NT));
    return (int)((tiles + 7) / 8);                    // 8 waves / block
  };

  // 0) weight prep + BN folding
  wt_convert<<<cd(128 * 64, 256),  256, 0, stream>>>(W2,   W2t,   64,  128);
  wt_convert<<<cd(256 * 128, 256), 256, 0, stream>>>(W3,   W3t,   128, 256);
  wt_convert<<<cd(256 * 256, 256), 256, 0, stream>>>(W4,   W4t,   256, 256);
  wt_convert<<<cd(256 * 256, 256), 256, 0, stream>>>(Wq,   Wqt,   256, 256);
  wt_convert<<<cd(256 * 256, 256), 256, 0, stream>>>(Wk,   Wkt,   256, 256);
  wt_convert<<<cd(256 * 256, 256), 256, 0, stream>>>(Wv,   Wvt,   256, 256);
  wt_convert<<<cd(256 * 256, 256), 256, 0, stream>>>(Wo,   Wot,   256, 256);
  wt_convert<<<cd(128 * 256, 256), 256, 0, stream>>>(W128, W128t, 256, 128);
  wt_convert<<<cd(16 * 128, 256),  256, 0, stream>>>(Wc,   Wct,   128, 16);
  bn_fold<<<1, 128, 0, stream>>>(bn2_g, bn2_b, bn2_m, bn2_v, sc2, sh2, 128);
  bn_fold<<<1, 256, 0, stream>>>(bn3_g, bn3_b, bn3_m, bn3_v, sc3, sh3, 256);

  // 1) PPmodel
  pp_stage1<<<cd(n, 256), 256, 0, stream>>>(pt_fea, bn0_g, bn0_b, bn0_m, bn0_v,
                                            W1, b1, bn1_g, bn1_b, bn1_m, bn1_v, h1, n);
  gemm_wmma_f16<64, 2, 2><<<gemmGrid(n, 128, 2, 2), 256, 0, stream>>>(
      h1, W2t, b2, sc2, sh2, nullptr, h2, n, 128, 1);
  gemm_wmma_f16<128, 2, 2><<<gemmGrid(n, 256, 2, 2), 256, 0, stream>>>(
      h2, W3t, b3, sc3, sh3, nullptr, h3, n, 256, 1);
  gemm_wmma_f16<256, 2, 2><<<gemmGrid(n, 256, 2, 2), 256, 0, stream>>>(
      h3, W4t, b4, nullptr, nullptr, mlp, mlp_h, n, 256, 0);

  // 2) MHA: q / k / v projections + L=4 softmax + output projection
  gemm_wmma_f16<256, 2, 2><<<gemmGrid(n, 256, 2, 2), 256, 0, stream>>>(
      mlp_h, Wqt, bq, nullptr, nullptr, qb, nullptr, n, 256, 0);
  cvt_f2h<<<cd(nL * LL * 256, 256), 256, 0, stream>>>(segfea, seg_h, nL * LL * 256);
  gemm_wmma_f16<256, 2, 2><<<gemmGrid(nL * LL, 256, 2, 2), 256, 0, stream>>>(
      seg_h, Wkt, bk, nullptr, nullptr, nullptr, k_h, n * LL, 256, 0);
  gemm_wmma_f16<256, 2, 2><<<gemmGrid(nL * LL, 256, 2, 2), 256, 0, stream>>>(
      seg_h, Wvt, bv, nullptr, nullptr, nullptr, v_h, n * LL, 256, 0);
  mha_attn<<<cd(nL * HH, 256), 256, 0, stream>>>(qb, k_h, v_h, attn_h, n);
  gemm_wmma_f16<256, 2, 2><<<gemmGrid(n, 256, 2, 2), 256, 0, stream>>>(
      attn_h, Wot, bo, nullptr, nullptr, sam, nullptr, n, 256, 0);

  // 3) unique + segment-max pooling
  hipMemsetAsync(counts, 0, sizeof(unsigned) * NKEYS, stream);
  hipMemsetAsync(slots, 0, sizeof(unsigned) * 3 * (size_t)NKEYS * DD, stream);
  pool_atomic<<<cd(nL * DD, 256), 256, 0, stream>>>(
      xy, mlp, sam, pixfea, slotM, slotS, slotP, counts, n);
  scan_ranks<<<1, 1024, 0, stream>>>(counts, rankArr, keyOfRank, nuniq);
  write_inv<<<cd(n, 256), 256, 0, stream>>>(xy, rankArr, o_inv, n);
  pool_gather<<<cd(nL * DD, 256), 256, 0, stream>>>(
      slotM, slotS, slotP, keyOfRank, nuniq, pmlp, pmlp_h, o_seg, o_pix, o_unq, n);

  // 4) to_128 + fea_compression
  gemm_wmma_f16<256, 2, 2><<<gemmGrid(n, 128, 2, 2), 256, 0, stream>>>(
      pmlp_h, W128t, b128, nullptr, nullptr, nullptr, ori_h, n, 128, 0);
  gemm_wmma_f16<128, 2, 1><<<gemmGrid(n, 16, 2, 1), 256, 0, stream>>>(
      ori_h, Wct, bc, nullptr, nullptr, o_proc, nullptr, n, 16, 1);
}
